// FocusCascadedBlock_16149077033368
// MI455X (gfx1250) — matrix operand order repaired and checked by reading, not compiled, not run on hardware
//
#include <hip/hip_runtime.h>
#include <hip/hip_bf16.h>
#include <math.h>

typedef __attribute__((ext_vector_type(16))) _Float16     v16h;
typedef __attribute__((ext_vector_type(8)))  float        v8f;
typedef __attribute__((ext_vector_type(4)))  unsigned int ui4;
typedef _Float16 h16;

#define BB     32
#define NVK    4096
#define DD     640
#define HH     8
#define DHD    80
#define TOPKK  256
#define CHUNKK 8
#define ADIMM  7
#define NAA    56
#define MA     (BB*NAA)       /* 1792  */
#define MV     (BB*NVK)       /* 131072*/
#define MAQ    (BB*CHUNKK)    /* 256   */
#define ROWSF  (BB*HH*NAA)    /* 14336 */
#define SCALE_ATTN 0.11180339887498948f  /* 1/sqrt(80) */

// ---------------- input leaf indices (JAX pytree sorted-key flatten) -------
enum {
  IN_A=0, IN_AQ=1, IN_V=2, IN_VMASK=3,
  AQ_BK=4, AQ_BO=5, AQ_BQ=6, AQ_BV=7, AQ_WK=8, AQ_WO=9, AQ_WQ=10, AQ_WV=11,
  FFN_B1=12, FFN_B2=13, FFN_LN_B=14, FFN_LN_G=15, FFN_W1=16, FFN_W2=17,
  FUS_B1=18, FUS_B2=19, FUS_W1=20, FUS_W2=21,
  LNAQK_B=22, LNAQK_G=23, LNAQQ_B=24, LNAQQ_G=25,
  LNFA_B=26, LNFA_G=27, LNFAQ_B=28, LNFAQ_G=29, LNFV_B=30, LNFV_G=31,
  LNSQ_B=32, LNSQ_G=33,
  SA_BK=34, SA_BO=35, SA_BQ=36, SA_BV=37, SA_WK=38, SA_WO=39, SA_WQ=40, SA_WV=41,
  V_BK=42, V_BO=43, V_BQ=44, V_BV=45, V_GB=46, V_GLN_B=47, V_GLN_G=48, V_GW=49,
  V_QLN_B=50, V_QLN_G=51, V_VLN_B=52, V_VLN_G=53, V_WK=54, V_WO=55, V_WQ=56, V_WV=57
};

// ------------------------------ reductions --------------------------------
__device__ __forceinline__ float blk_sum(float v, float* buf) {
  int t = threadIdx.x;
  buf[t] = v; __syncthreads();
  for (int s = blockDim.x >> 1; s > 0; s >>= 1) {
    if (t < s) buf[t] += buf[t + s];
    __syncthreads();
  }
  float r = buf[0]; __syncthreads();
  return r;
}
__device__ __forceinline__ float blk_max(float v, float* buf) {
  int t = threadIdx.x;
  buf[t] = v; __syncthreads();
  for (int s = blockDim.x >> 1; s > 0; s >>= 1) {
    if (t < s) buf[t] = fmaxf(buf[t], buf[t + s]);
    __syncthreads();
  }
  float r = buf[0]; __syncthreads();
  return r;
}

// ------------------------------ WMMA GEMM ---------------------------------
// Y[z][m,n] = act( X[z][m,:K] * W[z][n,:K]^T + bias[n] )
// W is ALWAYS column-major per output column n ("transposed" layout):
//   element (k,n) lives at W[n*ldw + k].
// z decomposes as (z/zdiv, z%zdiv) with (outer, inner) element strides.
struct GemmP {
  const h16* X; const h16* W; const float* bias;
  float* Yf; h16* Yh;
  int M, N, K, lda, ldw, ldc;
  long long sxo, sxi, swo, swi, syo, syi;
  int zdiv, act;
};

__global__ __launch_bounds__(256) void wmma_gemm(GemmP p) {
  const int lane = threadIdx.x & 31;
  const int wv   = threadIdx.x >> 5;
  const int n0   = (blockIdx.x * 8 + wv) * 16;
  const int m0   = blockIdx.y * 16;
  const int z    = blockIdx.z;
  if (n0 >= p.N) return;                     // wave-uniform exit, EXEC stays full

  const h16* __restrict__ X = p.X + (long long)(z / p.zdiv) * p.sxo
                                  + (long long)(z % p.zdiv) * p.sxi;
  const h16* __restrict__ W = p.W + (long long)(z / p.zdiv) * p.swo
                                  + (long long)(z % p.zdiv) * p.swi;
  const long long yb = (long long)(z / p.zdiv) * p.syo + (long long)(z % p.zdiv) * p.syi;

  const int g  = lane >> 4;                  // lane group 0/1
  int mA = m0 + (lane & 15); if (mA >= p.M) mA = p.M - 1;  // clamp; stores masked
  int nB = n0 + (lane & 15); if (nB >= p.N) nB = p.N - 1;
  const h16* __restrict__ Xr = X + (long long)mA * p.lda;
  const h16* __restrict__ Wr = W + (long long)nB * p.ldw;

  v8f acc = {};
  const int kfull = p.K & ~31;
  // ---- hot loop: 4x b128 loads + 1 wmma, no branches ----
  // A layout (16-bit 16x32): lane needs halves [k0+g*8, +8) and [k0+16+g*8, +8)
  // B layout (32x16):        lane needs halves [k0+g*8, +8) and [k0+16+g*8, +8)
  for (int k0 = 0; k0 < kfull; k0 += 32) {
    ui4 a0 = *(const ui4*)(Xr + k0 + g * 8);
    ui4 a1 = *(const ui4*)(Xr + k0 + g * 8 + 16);
    ui4 b0 = *(const ui4*)(Wr + k0 + g * 8);
    ui4 b1 = *(const ui4*)(Wr + k0 + g * 8 + 16);
    v16h a = __builtin_bit_cast(v16h, __builtin_shufflevector(a0, a1, 0,1,2,3,4,5,6,7));
    v16h b = __builtin_bit_cast(v16h, __builtin_shufflevector(b0, b1, 0,1,2,3,4,5,6,7));
    acc = __builtin_amdgcn_wmma_f32_16x16x32_f16(false, a, false, b,
                                                 (short)0, acc, false, false);
  }
  // ---- tail (K % 32): unconditional loads, cndmask zero-fill ----
  if (kfull < p.K) {
    const int k0 = kfull;
    v16h a, b;
#pragma unroll
    for (int j = 0; j < 8; ++j) {
      int ka = k0 + ((j < 4) ? 0 : 16) + g * 8 + (j & 3) * 2;
      h16 x0 = Xr[ka], x1 = Xr[ka + 1];
      a[2*j]   = (ka     < p.K) ? x0 : (h16)0.f;
      a[2*j+1] = (ka + 1 < p.K) ? x1 : (h16)0.f;
      int kb = k0 + ((j < 4) ? 0 : 16) + g * 8 + (j & 3) * 2;
      h16 w0 = Wr[kb], w1 = Wr[kb + 1];
      b[2*j]   = (kb     < p.K) ? w0 : (h16)0.f;
      b[2*j+1] = (kb + 1 < p.K) ? w1 : (h16)0.f;
    }
    acc = __builtin_amdgcn_wmma_f32_16x16x32_f16(false, a, false, b,
                                                 (short)0, acc, false, false);
  }

  const int nOut = n0 + (lane & 15);
#pragma unroll
  for (int r = 0; r < 8; ++r) {
    int m = m0 + r + (g << 3);               // D layout: VGPR r -> row r / r+8
    if (m < p.M && nOut < p.N) {
      float v = acc[r];
      if (p.bias) v += p.bias[nOut];
      if (p.act == 1)      v = 0.5f * v * (1.f + erff(v * 0.70710678118654752f)); // GELU
      else if (p.act == 2) v = 1.f / (1.f + __expf(-v));                           // sigmoid
      long long yi = yb + (long long)m * p.ldc + nOut;
      if (p.Yf) p.Yf[yi] = v;
      if (p.Yh) p.Yh[yi] = (h16)v;
    }
  }
}

// --------------------- convert + transpose (weights) ----------------------
// x: f32 [K,N] row-major  ->  y: f16 [N,K] row-major
__global__ void cvtt_k(const float* __restrict__ x, h16* __restrict__ y, int K, int N) {
  __shared__ float t[32][33];
  const int k0 = blockIdx.y * 32, n0 = blockIdx.x * 32;
  const int tx = threadIdx.x & 31, ty = threadIdx.x >> 5;
  for (int i = 0; i < 32; i += 8) {
    int k = k0 + ty + i, n = n0 + tx;
    t[ty + i][tx] = (k < K && n < N) ? x[(long long)k * N + n] : 0.f;
  }
  __syncthreads();
  for (int i = 0; i < 32; i += 8) {
    int n = n0 + ty + i, k = k0 + tx;
    if (n < N && k < K) y[(long long)n * K + k] = (h16)t[tx][ty + i];
  }
}

// --------------------- h16 transpose (per batch z) ------------------------
// x: [z][R,C] -> y: [z][C,R]
__global__ void tr_k(const h16* __restrict__ x, h16* __restrict__ y, int R, int C) {
  __shared__ h16 t[32][33];
  const long long zb = (long long)blockIdx.z * R * C;
  const int r0 = blockIdx.y * 32, c0 = blockIdx.x * 32;
  const int tx = threadIdx.x & 31, ty = threadIdx.x >> 5;
  for (int i = 0; i < 32; i += 8) {
    int r = r0 + ty + i, c = c0 + tx;
    t[ty + i][tx] = (r < R && c < C) ? x[zb + (long long)r * C + c] : (h16)0.f;
  }
  __syncthreads();
  for (int i = 0; i < 32; i += 8) {
    int c = c0 + ty + i, r = r0 + tx;
    if (c < C && r < R) y[zb + (long long)c * R + r] = t[tx][ty + i];
  }
}

// ------------------------------ LayerNorm ---------------------------------
__global__ void ln_kernel(const float* __restrict__ x, const float* __restrict__ g,
                          const float* __restrict__ b, h16* __restrict__ y,
                          int ldy, int coff) {
  __shared__ float buf[256];
  const long long row = blockIdx.x;
  const float* xr = x + row * DD;
  float s = 0.f;
  for (int i = threadIdx.x; i < DD; i += 256) s += xr[i];
  float mean = blk_sum(s, buf) * (1.f / DD);
  float v = 0.f;
  for (int i = threadIdx.x; i < DD; i += 256) { float d = xr[i] - mean; v += d * d; }
  float var = blk_sum(v, buf) * (1.f / DD);
  float rs = rsqrtf(var + 1e-5f);
  h16* yr = y + row * (long long)ldy + coff;
  for (int i = threadIdx.x; i < DD; i += 256)
    yr[i] = (h16)((xr[i] - mean) * rs * g[i] + b[i]);
}

// ------------------------- elementwise helpers ----------------------------
__global__ void add_k(const float* __restrict__ a, const float* __restrict__ b,
                      float* __restrict__ y, long long n) {
  long long i = (long long)blockIdx.x * blockDim.x + threadIdx.x;
  long long st = (long long)gridDim.x * blockDim.x;
  for (; i < n; i += st) y[i] = a[i] + b[i];
}
__global__ void mul_k(const float* __restrict__ a, const float* __restrict__ b,
                      float* __restrict__ y, long long n) {
  long long i = (long long)blockIdx.x * blockDim.x + threadIdx.x;
  long long st = (long long)gridDim.x * blockDim.x;
  for (; i < n; i += st) y[i] = a[i] * b[i];
}

// ------------------------- softmax kernels --------------------------------
__global__ void self_softmax(const float* __restrict__ sc, h16* __restrict__ pr) {
  __shared__ float buf[64];
  const int row = blockIdx.x;
  const int q = row % NAA;
  const int k = threadIdx.x;
  bool inb = (k < NAA);
  bool ok = inb && ((q / ADIMM) >= (k / ADIMM));
  float v = ok ? sc[(long long)row * NAA + k] * SCALE_ATTN : -1e30f;
  float mx = blk_max(v, buf);
  float e = inb ? __expf(v - mx) : 0.f;
  float Z = blk_sum(e, buf);
  if (inb) pr[(long long)row * NAA + k] = (h16)(e / Z);
}

__global__ void aq_softmax(const float* __restrict__ sc, h16* __restrict__ pr, int rows) {
  int r = blockIdx.x * blockDim.x + threadIdx.x;
  if (r >= rows) return;
  const float* s = sc + (long long)r * CHUNKK;
  float mx = -1e30f;
  for (int j = 0; j < CHUNKK; ++j) mx = fmaxf(mx, s[j] * SCALE_ATTN);
  float e[CHUNKK], Z = 0.f;
  for (int j = 0; j < CHUNKK; ++j) { e[j] = __expf(s[j] * SCALE_ATTN - mx); Z += e[j]; }
  for (int j = 0; j < CHUNKK; ++j) pr[(long long)r * CHUNKK + j] = (h16)(e[j] / Z);
}

// focus: per-row exact top-256 threshold (deterministic) + softmax + entropy
__global__ void focus_softmax(const h16* __restrict__ sc, h16* __restrict__ pr,
                              float* __restrict__ pent) {
  __shared__ float sv[NVK];
  __shared__ unsigned short sk[NVK];
  __shared__ unsigned char sf[NVK];
  __shared__ float buf[256];
  __shared__ int scnt;
  const long long row = blockIdx.x;
  const int tid = threadIdx.x;
  const h16* srow = sc + row * NVK;
  for (int i = tid; i < NVK; i += 256) {
    h16 h = srow[i];
    unsigned short u = __builtin_bit_cast(unsigned short, h);
    u = (u & 0x8000u) ? (unsigned short)(~u) : (unsigned short)(u | 0x8000u);
    sk[i] = u; sv[i] = (float)h;
  }
  __syncthreads();
  unsigned lo = 0, hi = 65536;
  while (lo + 1 < hi) {
    unsigned mid = (lo + hi) >> 1;
    if (tid == 0) scnt = 0;
    __syncthreads();
    int c = 0;
    for (int i = tid; i < NVK; i += 256) c += (sk[i] >= mid);
    atomicAdd(&scnt, c); __syncthreads();
    int tot = scnt; __syncthreads();
    if (tot >= TOPKK) lo = mid; else hi = mid;
  }
  const unsigned T = lo;
  if (tid == 0) scnt = 0;
  __syncthreads();
  { int c = 0; for (int i = tid; i < NVK; i += 256) c += (sk[i] > T);
    atomicAdd(&scnt, c); }
  __syncthreads();
  const int need = TOPKK - scnt;   // ties at T admitted by ascending index
  __syncthreads();
  for (int i = tid; i < NVK; i += 256) {
    unsigned short u = sk[i];
    int sel = 0;
    if (u > T) sel = 1;
    else if (u == T) { int r = 0; for (int j = 0; j < i; ++j) r += (sk[j] == T); sel = (r < need); }
    sf[i] = (unsigned char)sel;
  }
  __syncthreads();
  float mx = -1e30f;
  for (int i = tid; i < NVK; i += 256) if (sf[i]) mx = fmaxf(mx, sv[i]);
  mx = blk_max(mx, buf) * SCALE_ATTN;
  float zs = 0.f;
  for (int i = tid; i < NVK; i += 256) if (sf[i]) zs += __expf(sv[i] * SCALE_ATTN - mx);
  float Z = blk_sum(zs, buf);
  float ent = 0.f;
  h16* prow = pr + row * NVK;
  for (int i = tid; i < NVK; i += 256) {
    float p = sf[i] ? __expf(sv[i] * SCALE_ATTN - mx) / Z : 0.f;
    prow[i] = (h16)p;
    float pc = p > 1e-8f ? p : 1e-8f;          // torch clamp_min over full matrix
    ent += -pc * __logf(pc);
  }
  ent = blk_sum(ent, buf);
  if (tid == 0) pent[row] = ent;
}

// ------------------------------ statistics --------------------------------
__global__ void attn_stats(const h16* __restrict__ pr, float* __restrict__ pamax) {
  __shared__ float buf[256];
  const int b = blockIdx.x / NAA, q = blockIdx.x % NAA;
  float vmax = 0.f;
  for (int k = threadIdx.x; k < NVK; k += 256) {
    float s = 0.f;
    for (int h = 0; h < HH; ++h)
      s += (float)pr[((long long)(b * HH + h) * NAA + q) * NVK + k];
    vmax = fmaxf(vmax, s * (1.f / HH));
  }
  vmax = blk_max(vmax, buf);
  if (threadIdx.x == 0) pamax[blockIdx.x] = vmax;
}

__global__ void rownorm_k(const float* __restrict__ x, float* __restrict__ out) {
  __shared__ float buf[256];
  const long long row = blockIdx.x;
  float s = 0.f;
  for (int i = threadIdx.x; i < DD; i += 256) { float v = x[row * DD + i]; s += v * v; }
  s = blk_sum(s, buf);
  if (threadIdx.x == 0) out[row] = sqrtf(s);
}

__global__ void gatesum_k(const float* __restrict__ g, float* __restrict__ out, long long n) {
  __shared__ float buf[256];
  float s = 0.f;
  for (long long i = (long long)blockIdx.x * 256 + threadIdx.x; i < n;
       i += (long long)gridDim.x * 256) s += g[i];
  s = blk_sum(s, buf);
  if (threadIdx.x == 0) out[blockIdx.x] = s;
}

__global__ void finalize_k(const float* pnA, const float* pnAQ, const float* pnV,
                           const float* pnVr, const float* pamax, const float* pent,
                           const float* pgate, float* outsc) {
  __shared__ float buf[256];
  float s;
  s = 0.f; for (int i = threadIdx.x; i < MA; i += 256) s += pnA[i];
  float nA  = blk_sum(s, buf) / MA;
  s = 0.f; for (int i = threadIdx.x; i < MA; i += 256) s += pnAQ[i];
  float nAQ = blk_sum(s, buf) / MA;
  s = 0.f; for (int i = threadIdx.x; i < MA; i += 256) s += pnV[i];
  float nV  = blk_sum(s, buf) / MA;
  s = 0.f; for (int i = threadIdx.x; i < MA; i += 256) s += pnVr[i];
  float nVr = blk_sum(s, buf) / MA;
  s = 0.f; for (int i = threadIdx.x; i < MA; i += 256) s += pamax[i];
  float amax = blk_sum(s, buf) / MA;
  s = 0.f; for (int i = threadIdx.x; i < ROWSF; i += 256) s += pent[i];
  float entm = blk_sum(s, buf) / ROWSF;
  s = 0.f; for (int i = threadIdx.x; i < 256; i += 256) s += pgate[i];
  float gm = blk_sum(s, buf) / ((float)MA * DD);
  if (threadIdx.x == 0) {
    float tot = nA + nAQ + nV + 1e-8f, totr = nA + nAQ + nVr + 1e-8f;
    outsc[0] = nA;  outsc[1] = nAQ; outsc[2] = nV;  outsc[3] = nVr;
    outsc[4] = nA / tot; outsc[5] = nAQ / tot; outsc[6] = nV / tot; outsc[7] = nVr / totr;
    outsc[8] = amax; outsc[9] = entm / logf((float)TOPKK); outsc[10] = entm; outsc[11] = gm;
  }
}

// ------------------------------ host side ---------------------------------
static void launch_gemm(hipStream_t st, const h16* X, const h16* W, const float* bias,
                        float* Yf, h16* Yh, int M, int N, int K,
                        int lda, int ldw, int ldc,
                        long long sxo, long long sxi, long long swo, long long swi,
                        long long syo, long long syi,
                        int nb, int zdiv, int act) {
  GemmP p{X, W, bias, Yf, Yh, M, N, K, lda, ldw, ldc,
          sxo, sxi, swo, swi, syo, syi, zdiv, act};
  dim3 g((N + 127) / 128, (M + 15) / 16, nb);
  wmma_gemm<<<g, 256, 0, st>>>(p);
}

extern "C" void kernel_launch(void* const* d_in, const int* in_sizes, int n_in,
                              void* d_out, int out_size, void* d_ws, size_t ws_size,
                              hipStream_t stream) {
  (void)in_sizes; (void)n_in; (void)out_size; (void)ws_size;
  const float* A  = (const float*)d_in[IN_A];
  const float* AQ = (const float*)d_in[IN_AQ];
  const float* V  = (const float*)d_in[IN_V];
  auto F = [&](int i) { return (const float*)d_in[i]; };

  // --------- workspace carve-out (256B aligned chunks) ---------
  char* wsp = (char*)d_ws;
  size_t off = 0;
  auto alloc = [&](size_t nbytes) -> void* {
    void* p = wsp + off; off += (nbytes + 255) & ~(size_t)255; return p;
  };
  const long long DDxDD = (long long)DD * DD;
  // f16 weights, stored TRANSPOSED [N,K]
  h16 *wsq = (h16*)alloc(DDxDD*2), *wsk = (h16*)alloc(DDxDD*2),
      *wsv = (h16*)alloc(DDxDD*2), *wso = (h16*)alloc(DDxDD*2);
  h16 *waqq = (h16*)alloc(DDxDD*2), *waqk = (h16*)alloc(DDxDD*2),
      *waqv = (h16*)alloc(DDxDD*2), *waqo = (h16*)alloc(DDxDD*2);
  h16 *wvq = (h16*)alloc(DDxDD*2), *wvk = (h16*)alloc(DDxDD*2),
      *wvv = (h16*)alloc(DDxDD*2), *wvo = (h16*)alloc(DDxDD*2);
  h16 *wgw = (h16*)alloc(DDxDD*2);
  h16 *wf1 = (h16*)alloc(3*DDxDD*2), *wf2 = (h16*)alloc(DDxDD*2);
  h16 *wffn1 = (h16*)alloc(4*DDxDD*2), *wffn2 = (h16*)alloc(4*DDxDD*2);
  // LN'd activations (f16)
  const long long NAD = (long long)MA * DD;
  h16 *An_self = (h16*)alloc(NAD*2), *An_aqq = (h16*)alloc(NAD*2),
      *An_fq = (h16*)alloc(NAD*2),  *An_g = (h16*)alloc(NAD*2);
  h16 *AQn = (h16*)alloc((long long)MAQ*DD*2);
  h16 *Vn  = (h16*)alloc((long long)MV*DD*2);
  // projections
  h16 *Qs = (h16*)alloc(NAD*2), *Ks = (h16*)alloc(NAD*2), *Vs = (h16*)alloc(NAD*2);
  h16 *Vst = (h16*)alloc(NAD*2);
  h16 *Qaq = (h16*)alloc(NAD*2);
  h16 *Kaq = (h16*)alloc((long long)MAQ*DD*2), *Vaq = (h16*)alloc((long long)MAQ*DD*2);
  h16 *Vaqt = (h16*)alloc((long long)MAQ*DD*2);
  h16 *Qf = (h16*)alloc(NAD*2);
  h16 *Kf = (h16*)alloc((long long)MV*DD*2);
  h16 *Valf = (h16*)alloc((long long)MV*DD*2), *Valt = (h16*)alloc((long long)MV*DD*2);
  // scores / probs
  float *sc_self = (float*)alloc((long long)ROWSF*NAA*4);
  h16   *pr_self = (h16*)alloc((long long)ROWSF*NAA*2);
  float *sc_aq   = (float*)alloc((long long)ROWSF*CHUNKK*4);
  h16   *pr_aq   = (h16*)alloc((long long)ROWSF*CHUNKK*2);
  h16   *sc_f    = (h16*)alloc((long long)ROWSF*NVK*2);
  h16   *pr_f    = (h16*)alloc((long long)ROWSF*NVK*2);
  // attention outputs (heads interleaved)
  h16 *aV_self = (h16*)alloc(NAD*2), *aV_aq = (h16*)alloc(NAD*2), *aV_f = (h16*)alloc(NAD*2);
  float *H_A = (float*)alloc(NAD*4), *H_AQ = (float*)alloc(NAD*4);
  float *HVraw = (float*)alloc(NAD*4), *HV = (float*)alloc(NAD*4), *gbuf = (float*)alloc(NAD*4);
  // fusion / ffn
  h16 *fin = (h16*)alloc((long long)MA*3*DD*2);
  h16 *fus_h = (h16*)alloc(NAD*2);
  float *fused = (float*)alloc(NAD*4), *A1 = (float*)alloc(NAD*4);
  h16 *A1n = (h16*)alloc(NAD*2);
  h16 *ffn_h = (h16*)alloc((long long)MA*4*DD*2);
  float *ffn_out = (float*)alloc(NAD*4);
  // stats partials
  float *pnA = (float*)alloc(MA*4), *pnAQ = (float*)alloc(MA*4),
        *pnV = (float*)alloc(MA*4), *pnVr = (float*)alloc(MA*4);
  float *pamax = (float*)alloc(MA*4);
  float *pent  = (float*)alloc(ROWSF*4);
  float *pgate = (float*)alloc(256*4);

  float* outA2 = (float*)d_out;
  float* outsc = outA2 + (size_t)MA * DD;

  // --------- 1. weight convert + transpose f32[K,N] -> f16[N,K] ---------
  struct WC { int idx; h16* dst; int K, N; } tbl[] = {
    {SA_WQ, wsq, DD, DD}, {SA_WK, wsk, DD, DD}, {SA_WV, wsv, DD, DD}, {SA_WO, wso, DD, DD},
    {AQ_WQ, waqq, DD, DD}, {AQ_WK, waqk, DD, DD}, {AQ_WV, waqv, DD, DD}, {AQ_WO, waqo, DD, DD},
    {V_WQ, wvq, DD, DD}, {V_WK, wvk, DD, DD}, {V_WV, wvv, DD, DD}, {V_WO, wvo, DD, DD},
    {V_GW, wgw, DD, DD},
    {FUS_W1, wf1, 3*DD, DD}, {FUS_W2, wf2, DD, DD},
    {FFN_W1, wffn1, DD, 4*DD}, {FFN_W2, wffn2, 4*DD, DD},
  };
  for (auto& t : tbl)
    cvtt_k<<<dim3((t.N + 31) / 32, (t.K + 31) / 32), 256, 0, stream>>>(
        (const float*)d_in[t.idx], t.dst, t.K, t.N);

  // --------- 2. LayerNorms ---------
  ln_kernel<<<MA, 256, 0, stream>>>(A,  F(LNSQ_G),  F(LNSQ_B),  An_self, DD, 0);
  ln_kernel<<<MA, 256, 0, stream>>>(A,  F(LNAQQ_G), F(LNAQQ_B), An_aqq,  DD, 0);
  ln_kernel<<<MAQ,256, 0, stream>>>(AQ, F(LNAQK_G), F(LNAQK_B), AQn,     DD, 0);
  ln_kernel<<<MA, 256, 0, stream>>>(A,  F(V_QLN_G), F(V_QLN_B), An_fq,   DD, 0);
  ln_kernel<<<MV, 256, 0, stream>>>(V,  F(V_VLN_G), F(V_VLN_B), Vn,      DD, 0);
  ln_kernel<<<MA, 256, 0, stream>>>(A,  F(V_GLN_G), F(V_GLN_B), An_g,    DD, 0);

  // --------- 3. self-attention (chunk-causal) ---------
  launch_gemm(stream, An_self, wsq, F(SA_BQ), nullptr, Qs, MA, DD, DD, DD, DD, DD,
              0,0,0,0,0,0, 1,1,0);
  launch_gemm(stream, An_self, wsk, F(SA_BK), nullptr, Ks, MA, DD, DD, DD, DD, DD,
              0,0,0,0,0,0, 1,1,0);
  launch_gemm(stream, An_self, wsv, F(SA_BV), nullptr, Vs, MA, DD, DD, DD, DD, DD,
              0,0,0,0,0,0, 1,1,0);
  tr_k<<<dim3(20, 2, BB), 256, 0, stream>>>(Vs, Vst, NAA, DD);
  launch_gemm(stream, Qs, Ks, nullptr, sc_self, nullptr, NAA, NAA, DHD, DD, DD, NAA,
              (long long)NAA*DD, DHD, (long long)NAA*DD, DHD,
              (long long)HH*NAA*NAA, (long long)NAA*NAA, BB*HH, HH, 0);
  self_softmax<<<ROWSF, 64, 0, stream>>>(sc_self, pr_self);
  launch_gemm(stream, pr_self, Vst, nullptr, nullptr, aV_self, NAA, DHD, NAA, NAA, NAA, DD,
              (long long)HH*NAA*NAA, (long long)NAA*NAA, (long long)DD*NAA, (long long)DHD*NAA,
              (long long)NAA*DD, DHD, BB*HH, HH, 0);
  launch_gemm(stream, aV_self, wso, F(SA_BO), H_A, nullptr, MA, DD, DD, DD, DD, DD,
              0,0,0,0,0,0, 1,1,0);

  // --------- 4. A->AQ cross attention ---------
  launch_gemm(stream, An_aqq, waqq, F(AQ_BQ), nullptr, Qaq, MA, DD, DD, DD, DD, DD,
              0,0,0,0,0,0, 1,1,0);
  launch_gemm(stream, AQn, waqk, F(AQ_BK), nullptr, Kaq, MAQ, DD, DD, DD, DD, DD,
              0,0,0,0,0,0, 1,1,0);
  launch_gemm(stream, AQn, waqv, F(AQ_BV), nullptr, Vaq, MAQ, DD, DD, DD, DD, DD,
              0,0,0,0,0,0, 1,1,0);
  tr_k<<<dim3(20, 1, BB), 256, 0, stream>>>(Vaq, Vaqt, CHUNKK, DD);
  launch_gemm(stream, Qaq, Kaq, nullptr, sc_aq, nullptr, NAA, CHUNKK, DHD, DD, DD, CHUNKK,
              (long long)NAA*DD, DHD, (long long)CHUNKK*DD, DHD,
              (long long)HH*NAA*CHUNKK, (long long)NAA*CHUNKK, BB*HH, HH, 0);
  aq_softmax<<<(ROWSF + 255) / 256, 256, 0, stream>>>(sc_aq, pr_aq, ROWSF);
  launch_gemm(stream, pr_aq, Vaqt, nullptr, nullptr, aV_aq, NAA, DHD, CHUNKK, CHUNKK, CHUNKK, DD,
              (long long)HH*NAA*CHUNKK, (long long)NAA*CHUNKK,
              (long long)DD*CHUNKK, (long long)DHD*CHUNKK,
              (long long)NAA*DD, DHD, BB*HH, HH, 0);
  launch_gemm(stream, aV_aq, waqo, F(AQ_BO), H_AQ, nullptr, MA, DD, DD, DD, DD, DD,
              0,0,0,0,0,0, 1,1,0);

  // --------- 5. FocusAttention over V ---------
  launch_gemm(stream, An_fq, wvq, F(V_BQ), nullptr, Qf, MA, DD, DD, DD, DD, DD,
              0,0,0,0,0,0, 1,1,0);
  launch_gemm(stream, Vn, wvk, F(V_BK), nullptr, Kf, MV, DD, DD, DD, DD, DD,
              0,0,0,0,0,0, 1,1,0);
  launch_gemm(stream, Vn, wvv, F(V_BV), nullptr, Valf, MV, DD, DD, DD, DD, DD,
              0,0,0,0,0,0, 1,1,0);
  tr_k<<<dim3(20, 128, BB), 256, 0, stream>>>(Valf, Valt, NVK, DD);
  launch_gemm(stream, Qf, Kf, nullptr, nullptr, sc_f, NAA, NVK, DHD, DD, DD, NVK,
              (long long)NAA*DD, DHD, (long long)NVK*DD, DHD,
              (long long)HH*NAA*NVK, (long long)NAA*NVK, BB*HH, HH, 0);
  focus_softmax<<<ROWSF, 256, 0, stream>>>(sc_f, pr_f, pent);
  launch_gemm(stream, pr_f, Valt, nullptr, nullptr, aV_f, NAA, DHD, NVK, NVK, NVK, DD,
              (long long)HH*NAA*NVK, (long long)NAA*NVK,
              (long long)DD*NVK, (long long)DHD*NVK,
              (long long)NAA*DD, DHD, BB*HH, HH, 0);
  launch_gemm(stream, aV_f, wvo, F(V_BO), HVraw, nullptr, MA, DD, DD, DD, DD, DD,
              0,0,0,0,0,0, 1,1,0);
  launch_gemm(stream, An_g, wgw, F(V_GB), gbuf, nullptr, MA, DD, DD, DD, DD, DD,
              0,0,0,0,0,0, 1,1,2);                         // sigmoid gate
  mul_k<<<1024, 256, 0, stream>>>(HVraw, gbuf, HV, NAD);

  // --------- 6. statistics ---------
  rownorm_k<<<MA, 256, 0, stream>>>(H_A,   pnA);
  rownorm_k<<<MA, 256, 0, stream>>>(H_AQ,  pnAQ);
  rownorm_k<<<MA, 256, 0, stream>>>(HV,    pnV);
  rownorm_k<<<MA, 256, 0, stream>>>(HVraw, pnVr);
  attn_stats<<<MA, 256, 0, stream>>>(pr_f, pamax);
  gatesum_k<<<256, 256, 0, stream>>>(gbuf, pgate, NAD);

  // --------- 7. fusion MLP + residual ---------
  ln_kernel<<<MA, 256, 0, stream>>>(H_A,  F(LNFA_G),  F(LNFA_B),  fin, 3*DD, 0);
  ln_kernel<<<MA, 256, 0, stream>>>(H_AQ, F(LNFAQ_G), F(LNFAQ_B), fin, 3*DD, DD);
  ln_kernel<<<MA, 256, 0, stream>>>(HV,   F(LNFV_G),  F(LNFV_B),  fin, 3*DD, 2*DD);
  launch_gemm(stream, fin, wf1, F(FUS_B1), nullptr, fus_h, MA, DD, 3*DD, 3*DD, 3*DD, DD,
              0,0,0,0,0,0, 1,1,1);                         // GELU
  launch_gemm(stream, fus_h, wf2, F(FUS_B2), fused, nullptr, MA, DD, DD, DD, DD, DD,
              0,0,0,0,0,0, 1,1,0);
  add_k<<<1024, 256, 0, stream>>>(A, fused, A1, NAD);

  // --------- 8. FFN + residual -> d_out ---------
  ln_kernel<<<MA, 256, 0, stream>>>(A1, F(FFN_LN_G), F(FFN_LN_B), A1n, DD, 0);
  launch_gemm(stream, A1n, wffn1, F(FFN_B1), nullptr, ffn_h, MA, 4*DD, DD, DD, DD, 4*DD,
              0,0,0,0,0,0, 1,1,1);                         // GELU
  launch_gemm(stream, ffn_h, wffn2, F(FFN_B2), ffn_out, nullptr, MA, DD, 4*DD, 4*DD, 4*DD, DD,
              0,0,0,0,0,0, 1,1,0);
  add_k<<<1024, 256, 0, stream>>>(A1, ffn_out, outA2, NAD);

  // --------- 9. scalar outputs ---------
  finalize_k<<<1, 256, 0, stream>>>(pnA, pnAQ, pnV, pnVr, pamax, pent, pgate, outsc);
}
